// Model_43069932045089
// MI455X (gfx1250) — matrix operand order repaired and checked by reading, not compiled, not run on hardware
//
#include <hip/hip_runtime.h>
#include <hip/hip_bf16.h>

typedef __attribute__((ext_vector_type(16))) _Float16 v16h;
typedef __attribute__((ext_vector_type(8)))  _Float16 v8h;
typedef __attribute__((ext_vector_type(8)))  float    v8f;

#define NP    256   // padded node count
#define NV    255   // valid node count
#define BFULL 512   // full batch
#define BC    128   // batch chunk
#define MC    (BC * NP)   // rows per chunk (32768)
#define NNZC  2550

#define WMMA_F16(a, b, c) \
    __builtin_amdgcn_wmma_f32_16x16x32_f16(false, (a), false, (b), (short)0, (c), false, false)

// ---------------------------------------------------------------------------
// WMMA fragment loaders (contiguous k = 16h + s order, wave32)
// ---------------------------------------------------------------------------

// A: 16x32 f16 from row-major source, leading dim lda (halves).
// lane L: row = L&15, hi = L>>4; K halves 16hi..16hi+15
// -> one contiguous 32B chunk.
__device__ __forceinline__ v16h frag_a(const _Float16* __restrict__ A, int lda,
                                       int rowBase, int kBase, int lane)
{
    int r  = lane & 15;
    int hi = lane >> 4;
    const _Float16* row = A + (size_t)(rowBase + r) * lda + kBase + 16 * hi;
    v16h a;
#pragma unroll
    for (int v = 0; v < 8; ++v) {
        int k = 2 * v;
        a[2 * v]     = row[k];
        a[2 * v + 1] = row[k + 1];
    }
    return a;
}

// B: 32x16 f16 taken from a TRANSPOSED source BT [N x K], leading dim ldt.
// lane L: col = L&15; lanes 0-15 hold K=kBase..+15, lanes 16-31 K=kBase+16..+31,
// linearly across VGPRs -> one contiguous 32B (aligned) vector load.
__device__ __forceinline__ v16h frag_bt(const _Float16* __restrict__ BT, int ldt,
                                        int kBase, int colBase, int lane)
{
    int c  = lane & 15;
    int hi = lane >> 4;
    const _Float16* p = BT + (size_t)(colBase + c) * ldt + kBase + 16 * hi;
    return *(const v16h*)p;
}

// ---------------------------------------------------------------------------
// Prep kernels
// ---------------------------------------------------------------------------

__global__ void k_zero_f32(float* __restrict__ p, int n)
{
    int i = blockIdx.x * blockDim.x + threadIdx.x;
    if (i < n) p[i] = 0.0f;
}

__global__ void k_scatter_coo(const int* __restrict__ idx, const float* __restrict__ val,
                              float* __restrict__ D, int nnz)
{
    int i = blockIdx.x * blockDim.x + threadIdx.x;
    if (i < nnz) atomicAdd(&D[idx[2 * i] * NP + idx[2 * i + 1]], val[i]);
}

__global__ void k_f32_to_f16(const float* __restrict__ s, _Float16* __restrict__ d, int n)
{
    int i = blockIdx.x * blockDim.x + threadIdx.x;
    if (i < n) d[i] = (_Float16)s[i];
}

// W [din x dout] f32 -> WT [doutP x dinP] f16, zero padded (transposed!)
__global__ void k_pad_weight_t(const float* __restrict__ W, _Float16* __restrict__ WT,
                               int din, int dout, int dinP, int doutP)
{
    int i = blockIdx.x * blockDim.x + threadIdx.x;
    if (i >= dinP * doutP) return;
    int n = i / dinP, r = i % dinP;
    WT[i] = (r < din && n < dout) ? (_Float16)W[r * dout + n] : (_Float16)0.0f;
}

__global__ void k_pad_bias(const float* __restrict__ b, float* __restrict__ bp,
                           int dout, int doutP)
{
    int i = blockIdx.x * blockDim.x + threadIdx.x;
    if (i < doutP) bp[i] = (i < dout) ? b[i] : 0.0f;
}

// H [512,255,2] f32 -> X0 [512,256,32] f16 zero-padded (row-major)
__global__ void k_pack_input(const float* __restrict__ H, _Float16* __restrict__ X0, int n)
{
    int i = blockIdx.x * blockDim.x + threadIdx.x;
    if (i >= n) return;
    int c = i & 31;
    int r = (i >> 5) & 255;
    int b = i >> 13;
    X0[i] = (r < NV && c < 2) ? (_Float16)H[((size_t)b * NV + r) * 2 + c] : (_Float16)0.0f;
}

// ---------------------------------------------------------------------------
// Stage 1: YT[Nn x M] = (X[M x K] @ W[K x Nn] + bias)^T
// A = X row-major, B = WT [Nn x K].  Wave tile: 32x32 (2x2 WMMA blocking).
// grid = (Nn/32, M/256), block = 256 (8 waves)
// ---------------------------------------------------------------------------
__global__ void __launch_bounds__(256)
k_gemm_bias(const _Float16* __restrict__ X, const _Float16* __restrict__ WT,
            const float* __restrict__ bias, _Float16* __restrict__ YT,
            int K, int Nn)
{
    int lane    = threadIdx.x & 31;
    int wave    = threadIdx.x >> 5;
    int rowBase = blockIdx.y * 256 + wave * 32;
    int colBase = blockIdx.x * 32;

    v8f acc00 = {}, acc01 = {}, acc10 = {}, acc11 = {};
    for (int k = 0; k < K; k += 32) {
        v16h a0 = frag_a(X, K, rowBase, k, lane);
        v16h a1 = frag_a(X, K, rowBase + 16, k, lane);
        v16h b0 = frag_bt(WT, K, k, colBase, lane);
        v16h b1 = frag_bt(WT, K, k, colBase + 16, lane);
        acc00 = WMMA_F16(a0, b0, acc00);
        acc01 = WMMA_F16(a0, b1, acc01);
        acc10 = WMMA_F16(a1, b0, acc10);
        acc11 = WMMA_F16(a1, b1, acc11);
    }

    int hi = lane >> 4;
    int c  = lane & 15;
    float bv0 = bias[colBase + c];
    float bv1 = bias[colBase + 16 + c];

    // transposed store: each acc -> 8 consecutive rows of one column = 16B store
    auto st = [&](const v8f& acc, int n, int m0, float bv) {
        v8h y;
#pragma unroll
        for (int j = 0; j < 8; ++j) y[j] = (_Float16)(acc[j] + bv);
        *(v8h*)(YT + (size_t)n * MC + m0) = y;
    };
    int m0 = rowBase + 8 * hi;
    st(acc00, colBase + c,      m0,      bv0);
    st(acc01, colBase + 16 + c, m0,      bv1);
    st(acc10, colBase + c,      m0 + 16, bv0);
    st(acc11, colBase + 16 + c, m0 + 16, bv1);
}

// ---------------------------------------------------------------------------
// Stage 2: per batch element  Z_b = ReLU(DAD[256x256] @ Y_b[256 x Nn])
// A = DAD row-major, B = YT [Nn x M] (columns bb*256..+255).
// Z stored row-major (it is the next layer's A operand).
// grid = (Nn/32, BC), block = 256 (8 waves x 32 rows = M of 256)
// ---------------------------------------------------------------------------
__global__ void __launch_bounds__(256)
k_gemm_dad(const _Float16* __restrict__ DAD, const _Float16* __restrict__ YT,
           _Float16* __restrict__ Z, int Nn, float* __restrict__ outF32, int batchBase)
{
    int lane    = threadIdx.x & 31;
    int wave    = threadIdx.x >> 5;
    int rowBase = wave * 32;
    int colBase = blockIdx.x * 32;
    int bb      = blockIdx.y;
    int kOff    = bb * NP;

    v8f acc00 = {}, acc01 = {}, acc10 = {}, acc11 = {};
#pragma unroll
    for (int k = 0; k < NP; k += 32) {
        v16h a0 = frag_a(DAD, NP, rowBase, k, lane);
        v16h a1 = frag_a(DAD, NP, rowBase + 16, k, lane);
        v16h b0 = frag_bt(YT, MC, kOff + k, colBase, lane);
        v16h b1 = frag_bt(YT, MC, kOff + k, colBase + 16, lane);
        acc00 = WMMA_F16(a0, b0, acc00);
        acc01 = WMMA_F16(a0, b1, acc01);
        acc10 = WMMA_F16(a1, b0, acc10);
        acc11 = WMMA_F16(a1, b1, acc11);
    }

    int hi = lane >> 4;
    int c  = lane & 15;
    int r0 = rowBase + 8 * hi;

    if (outF32) {
        auto sto = [&](const v8f& acc, int rb, int n) {
#pragma unroll
            for (int j = 0; j < 8; ++j) {
                int r = rb + j;
                if (r < NV && n < 2)
                    outF32[((size_t)(batchBase + bb) * NV + r) * 2 + n] = fmaxf(acc[j], 0.0f);
            }
        };
        sto(acc00, r0,      colBase + c);
        sto(acc01, r0,      colBase + 16 + c);
        sto(acc10, r0 + 16, colBase + c);
        sto(acc11, r0 + 16, colBase + 16 + c);
    } else {
        _Float16* Zb = Z + (size_t)bb * NP * Nn;
        auto stz = [&](const v8f& acc, int rb, int n) {
#pragma unroll
            for (int j = 0; j < 8; ++j)
                Zb[(size_t)(rb + j) * Nn + n] = (_Float16)fmaxf(acc[j], 0.0f);
        };
        stz(acc00, r0,      colBase + c);
        stz(acc01, r0,      colBase + 16 + c);
        stz(acc10, r0 + 16, colBase + c);
        stz(acc11, r0 + 16, colBase + 16 + c);
    }
}

// ---------------------------------------------------------------------------
// Host orchestration
// ---------------------------------------------------------------------------
extern "C" void kernel_launch(void* const* d_in, const int* in_sizes, int n_in,
                              void* d_out, int out_size, void* d_ws, size_t ws_size,
                              hipStream_t stream)
{
    (void)in_sizes; (void)n_in; (void)out_size; (void)ws_size;

    const float* H       = (const float*)d_in[0];
    const float* sm_vals = (const float*)d_in[1];
    const float* sp_vals = (const float*)d_in[2];
    const float* Wf[6];
    const float* bf[6];
    for (int i = 0; i < 6; ++i) {
        Wf[i] = (const float*)d_in[3 + 2 * i];
        bf[i] = (const float*)d_in[4 + 2 * i];
    }
    const int* sm_idx = (const int*)d_in[15];
    const int* sp_idx = (const int*)d_in[16];

    static const int dims[7] = {2, 400, 300, 100, 300, 400, 2};
    static const int P[7]    = {32, 416, 320, 128, 320, 416, 32};

    // workspace carve-out (~64 MB total), 256B aligned
    char* w = (char*)d_ws;
    auto take = [&](size_t bytes) -> char* {
        char* p = w;
        w += (bytes + 255) & ~(size_t)255;
        return p;
    };
    float*    DADsm32 = (float*)take((size_t)NP * NP * 4);
    float*    DADsp32 = (float*)take((size_t)NP * NP * 4);
    _Float16* DADsm16 = (_Float16*)take((size_t)NP * NP * 2);
    _Float16* DADsp16 = (_Float16*)take((size_t)NP * NP * 2);
    _Float16* X0      = (_Float16*)take((size_t)BFULL * NP * P[0] * 2);
    _Float16* bufA    = (_Float16*)take((size_t)MC * 416 * 2);   // YT ping
    _Float16* bufB    = (_Float16*)take((size_t)MC * 416 * 2);   // Z  pong
    _Float16* WT[6];
    float*    bp[6];
    for (int i = 0; i < 6; ++i) {
        WT[i] = (_Float16*)take((size_t)P[i] * P[i + 1] * 2);
        bp[i] = (float*)take((size_t)P[i + 1] * 4);
    }

    // densify both Laplacians (zero -> atomic scatter -> f16)
    k_zero_f32<<<(NP * NP + 255) / 256, 256, 0, stream>>>(DADsm32, NP * NP);
    k_zero_f32<<<(NP * NP + 255) / 256, 256, 0, stream>>>(DADsp32, NP * NP);
    k_scatter_coo<<<(NNZC + 255) / 256, 256, 0, stream>>>(sm_idx, sm_vals, DADsm32, NNZC);
    k_scatter_coo<<<(NNZC + 255) / 256, 256, 0, stream>>>(sp_idx, sp_vals, DADsp32, NNZC);
    k_f32_to_f16<<<(NP * NP + 255) / 256, 256, 0, stream>>>(DADsm32, DADsm16, NP * NP);
    k_f32_to_f16<<<(NP * NP + 255) / 256, 256, 0, stream>>>(DADsp32, DADsp16, NP * NP);

    // pad/convert weights (transposed) and biases
    for (int i = 0; i < 6; ++i) {
        int n = P[i] * P[i + 1];
        k_pad_weight_t<<<(n + 255) / 256, 256, 0, stream>>>(Wf[i], WT[i],
                                                            dims[i], dims[i + 1], P[i], P[i + 1]);
        k_pad_bias<<<(P[i + 1] + 255) / 256, 256, 0, stream>>>(bf[i], bp[i],
                                                               dims[i + 1], P[i + 1]);
    }

    // pack input H -> padded f16 X0 (row-major)
    {
        int n = BFULL * NP * P[0];
        k_pack_input<<<(n + 255) / 256, 256, 0, stream>>>(H, X0, n);
    }

    // 6 layers, chunked over batch so activations stay L2-resident
    for (int c = 0; c < BFULL / BC; ++c) {
        const _Float16* Xc = X0 + (size_t)c * BC * NP * P[0];
        for (int L = 0; L < 6; ++L) {
            int K  = P[L];
            int Nn = P[L + 1];
            const _Float16* Xin = (L == 0) ? Xc : (const _Float16*)bufB;

            dim3 g1(Nn / 32, MC / 256);
            k_gemm_bias<<<g1, 256, 0, stream>>>(Xin, WT[L], bp[L], bufA, K, Nn);

            dim3 g2(Nn / 32, BC);
            k_gemm_dad<<<g2, 256, 0, stream>>>((L < 3) ? DADsm16 : DADsp16, bufA, bufB, Nn,
                                               (L == 5) ? (float*)d_out : nullptr, c * BC);
        }
    }
}
